// GCNLayer_6038724019025
// MI455X (gfx1250) — matrix operand order repaired and checked by reading, not compiled, not run on hardware
//
#include <hip/hip_runtime.h>
#include <hip/hip_bf16.h>

// GCN layer: h = x @ W (bf16 WMMA, f32 accum) ; out = relu(segment_sum(w_e * h[src], dst))
// x: [50000, 512] f32, W: [512, 256] f32, edges: 800000, out: [50000, 256] f32
// d_ws holds h: 50000*256*4 = 51.2 MB

#define GCN_NIN  512
#define GCN_NOUT 256

typedef __attribute__((ext_vector_type(16))) __bf16 v16bf;
typedef __attribute__((ext_vector_type(8)))  __bf16 v8bf;
typedef __attribute__((ext_vector_type(8)))  float  v8f;

// ---------------------------------------------------------------------------
// Kernel 1: h = x @ W using v_wmma_f32_16x16x32_bf16.
// Block = 256 threads = 8 waves. Block tile: M=64 rows x N=256 cols.
// Wave w: rows (w&3)*16 .. +15, cols (w>>2)*128 .. +127  -> 8 WMMA tiles,
// one shared A fragment per K-step. W slab (32K x 256N) staged transposed
// in LDS as bf16 (16 KB). All 8 B fragments are preloaded into distinct
// registers so the 8 WMMAs issue back-to-back instead of stalling on LDS
// latency between each one.
// ---------------------------------------------------------------------------
__global__ __launch_bounds__(256) void gcn_gemm_wmma(
    const float* __restrict__ x, const float* __restrict__ W,
    float* __restrict__ h, int nNodes)
{
    __shared__ __bf16 WT[GCN_NOUT * 32];   // WT[n][ki] = W[k0+ki][n], 16 KB

    const int tid     = threadIdx.x;
    const int lane    = tid & 31;
    const int wave    = tid >> 5;          // 0..7
    const int lsub    = lane & 15;         // 0..15
    const int halfSel = lane >> 4;         // 0 or 1

    const int mBase = blockIdx.x * 64;
    const int mSub  = (wave & 3) * 16;     // wave's row sub-tile
    const int nGrp  = (wave >> 2) * 128;   // wave's first column (8 tiles of 16)

    // A-row for this lane; clamp for the tail block (stores are guarded below)
    int rowA = mBase + mSub + lsub;
    if (rowA >= nNodes) rowA = nNodes - 1;
    const float* xrow = x + (size_t)rowA * GCN_NIN;

    v8f acc[8] = {};

    for (int k0 = 0; k0 < GCN_NIN; k0 += 32) {
        // ---- stage transposed bf16 W slab: thread t owns column n = t ----
        __syncthreads();
        #pragma unroll
        for (int ki = 0; ki < 32; ++ki) {
            WT[tid * 32 + ki] = (__bf16)W[(k0 + ki) * GCN_NOUT + tid];
        }
        __syncthreads();

        // prefetch next K-slab of x (global_prefetch_b8) to overlap with WMMAs
        if (k0 + 32 < GCN_NIN) {
            __builtin_prefetch(xrow + k0 + 32 + halfSel * 8, 0, 1);
        }

        // ---- A fragment (16x32 bf16), ISA wave32 layout:
        //   lanes 0-15:  M=lane,    elems 0-7 -> K=k0+0..7,  8-15 -> K=k0+16..23
        //   lanes 16-31: M=lane-16, elems 0-7 -> K=k0+8..15, 8-15 -> K=k0+24..31
        const float* xr = xrow + k0 + halfSel * 8;
        v16bf afrag;
        #pragma unroll
        for (int i = 0; i < 8; ++i) {
            afrag[i]     = (__bf16)xr[i];
            afrag[i + 8] = (__bf16)xr[16 + i];
        }

        // ---- preload all 8 B fragments (16 clause-able ds_load_b128) ----
        v16bf bfrag[8];
        #pragma unroll
        for (int t = 0; t < 8; ++t) {
            const __bf16* bp = &WT[(nGrp + t * 16 + lsub) * 32 + halfSel * 8];
            v8bf blo = *(const v8bf*)(bp);
            v8bf bhi = *(const v8bf*)(bp + 16);
            bfrag[t] = __builtin_shufflevector(blo, bhi,
                0, 1, 2, 3, 4, 5, 6, 7, 8, 9, 10, 11, 12, 13, 14, 15);
        }

        // ---- 8 back-to-back WMMAs sharing the A fragment ----
        #pragma unroll
        for (int t = 0; t < 8; ++t) {
            acc[t] = __builtin_amdgcn_wmma_f32_16x16x32_bf16(
                         false, afrag, false, bfrag[t], (short)0, acc[t],
                         false, false);
        }
    }

    // ---- store D: elem j of lane -> row = mBase+mSub + j + halfSel*8,
    //               col = nGrp + t*16 + lsub
    float* hp = h + (size_t)(mBase + mSub + halfSel * 8) * GCN_NOUT + nGrp + lsub;
    if (mBase + 64 <= nNodes) {
        // full block: branch-free clause'd stores
        #pragma unroll
        for (int t = 0; t < 8; ++t) {
            #pragma unroll
            for (int j = 0; j < 8; ++j) {
                hp[(size_t)j * GCN_NOUT + t * 16] = acc[t][j];
            }
        }
    } else {
        #pragma unroll
        for (int t = 0; t < 8; ++t) {
            #pragma unroll
            for (int j = 0; j < 8; ++j) {
                const int row = mBase + mSub + j + halfSel * 8;
                if (row < nNodes) {
                    hp[(size_t)j * GCN_NOUT + t * 16] = acc[t][j];
                }
            }
        }
    }
}

// ---------------------------------------------------------------------------
// Kernel 2: edge scatter. One wave32 per edge; lane handles 8 contiguous cols.
// h[src] gathers hit L2 (h = 51 MB < 192 MB L2); accumulation via native
// global_atomic_add_f32.
// ---------------------------------------------------------------------------
__global__ __launch_bounds__(256) void gcn_scatter(
    const float* __restrict__ h,
    const int* __restrict__ esrc, const int* __restrict__ edst,
    const float* __restrict__ ew,
    float* out, int nEdges)
{
    const int lane = threadIdx.x & 31;
    const int edge = (int)((blockIdx.x * (unsigned)blockDim.x + threadIdx.x) >> 5);
    if (edge >= nEdges) return;

    const int   s = esrc[edge];
    const int   d = edst[edge];
    const float w = ew[edge];

    const float4* hs = (const float4*)(h + (size_t)s * GCN_NOUT + lane * 8);
    const float4 a = hs[0];
    const float4 b = hs[1];

    float* op = out + (size_t)d * GCN_NOUT + lane * 8;
    unsafeAtomicAdd(op + 0, w * a.x);
    unsafeAtomicAdd(op + 1, w * a.y);
    unsafeAtomicAdd(op + 2, w * a.z);
    unsafeAtomicAdd(op + 3, w * a.w);
    unsafeAtomicAdd(op + 4, w * b.x);
    unsafeAtomicAdd(op + 5, w * b.y);
    unsafeAtomicAdd(op + 6, w * b.z);
    unsafeAtomicAdd(op + 7, w * b.w);
}

// ---------------------------------------------------------------------------
// Kernel 3: in-place ReLU on the aggregated output.
// ---------------------------------------------------------------------------
__global__ __launch_bounds__(256) void gcn_relu(float* out, int n)
{
    const int i = blockIdx.x * blockDim.x + threadIdx.x;
    if (i < n) out[i] = fmaxf(out[i], 0.0f);
}

// ---------------------------------------------------------------------------
extern "C" void kernel_launch(void* const* d_in, const int* in_sizes, int n_in,
                              void* d_out, int out_size, void* d_ws, size_t ws_size,
                              hipStream_t stream)
{
    const float* x    = (const float*)d_in[0];
    const float* W    = (const float*)d_in[1];
    const int*   esrc = (const int*)d_in[2];
    const int*   edst = (const int*)d_in[3];
    const float* ew   = (const float*)d_in[4];
    float*       out  = (float*)d_out;
    float*       h    = (float*)d_ws;     // 50000*256*4 B = 51.2 MB scratch

    const int nNodes = in_sizes[0] / GCN_NIN;   // 50000
    const int nEdges = in_sizes[2];             // 800000
    const int total  = nNodes * GCN_NOUT;

    // 1) zero the segment-sum accumulator (d_out is poisoned by the harness)
    hipMemsetAsync(out, 0, (size_t)total * sizeof(float), stream);

    // 2) h = x @ W  (bf16 WMMA, f32 accum); 64 rows x 256 cols per block
    const int mTiles = (nNodes + 63) / 64;      // 782
    gcn_gemm_wmma<<<mTiles, 256, 0, stream>>>(x, W, h, nNodes);

    // 3) scatter-add edges: 8 edges (waves) per 256-thread block
    const int scatterBlocks = (nEdges + 7) / 8; // 100000
    gcn_scatter<<<scatterBlocks, 256, 0, stream>>>(h, esrc, edst, ew, out, nEdges);

    // 4) ReLU in place
    gcn_relu<<<(total + 255) / 256, 256, 0, stream>>>(out, total);
}
